// HPHA_11785390261000
// MI455X (gfx1250) — matrix operand order repaired and checked by reading, not compiled, not run on hardware
//
#include <hip/hip_runtime.h>
#include <math.h>

// ---------- types ----------
typedef unsigned short u16;
typedef __attribute__((ext_vector_type(2)))  u16          u16x2;
typedef __attribute__((ext_vector_type(4)))  unsigned int u32x4;
typedef __attribute__((ext_vector_type(16))) __bf16       v16bf;
typedef __attribute__((ext_vector_type(8)))  float        v8f;

union FragBF  { v16bf v; u32x4 q[2]; };
union BStage  { u16x2 h[8]; u32x4 q[2]; };   // two-halves staging: loads -> d16/d16_hi, no pack VALU

__device__ __forceinline__ u16 f2bf(float f) {
  unsigned int u = __float_as_uint(f);
  unsigned int r = 0x7FFFu + ((u >> 16) & 1u);
  return (u16)((u + r) >> 16);
}
__device__ __forceinline__ float bf2f(u16 v) {
  return __uint_as_float(((unsigned int)v) << 16);
}

#define LDSS 40   // padded LDS row stride (bf16 elems): 80B keeps 16B alignment, dodges bank conflicts

// =====================================================================
// Implicit-GEMM conv with bf16 WMMA over ZERO-HALO-PADDED inputs:
// the im2col gather is unconditional (no compares/selects, loads have no
// VALU consumers until commit). Compile-time KHW/KW/S; power-of-two
// output widths passed as shifts. Software-pipelined, double-buffered
// LDS, one barrier per K-step.
//  Block: 128 thr (4 waves); block tile M=64, N=64; each wave: 16(M)x64(N).
//  inPlane/inW: padded input plane stride / row stride.
//  outB (optional) may itself be halo-padded: obPlane/obW/obOff.
// =====================================================================
template<int KHW, int KW, int S>
__global__ __launch_bounds__(128)
void conv_gemm_wmma(const u16* __restrict__ Awt, const u16* __restrict__ Bin,
                    const float* __restrict__ bias,
                    float* __restrict__ outF, u16* __restrict__ outB,
                    int M, int K, int N,
                    int inPlane, int inW, int owShift,
                    int obPlane, int obW, int obOff, int relu)
{
  __shared__ __align__(16) u16 As[2][64 * LDSS];
  __shared__ __align__(16) u16 Bs[2][64 * LDSS];

  const int t    = threadIdx.x;
  const int lane = t & 31;
  const int wave = t >> 5;
  const int n0   = blockIdx.x * 64;
  const int m0   = blockIdx.y * 64;

  // B-gather assignment: thread covers (n_local, 16 consecutive kk)
  const int n_local = t & 63;
  const int kkh     = (t >> 6) * 16;
  const int n  = n0 + n_local;
  const int oy = n >> owShift;
  const int ox = n & ((1 << owShift) - 1);
  const int gbase = (oy * S) * inW + ox * S;   // padded-coord base (halo == conv pad)

  // A-load assignment: thread covers half a 32-wide row
  const int ar = t >> 1;
  const int ac = (t & 1) * 16;
  const unsigned arow = (unsigned)(m0 + ar) * (unsigned)K + (unsigned)ac;

  v8f acc[4];
  for (int j = 0; j < 4; ++j)
    for (int e = 0; e < 8; ++e) acc[j][e] = 0.f;

  u32x4  a0, a1;       // staged A half-row
  BStage bs;           // staged B gather (halves written directly by loads)

  auto stage = [&](int k0) {
    const u32x4* src = (const u32x4*)(Awt + arow + (unsigned)k0);
    a0 = src[0];
    a1 = src[1];
    #pragma unroll
    for (int u = 0; u < 16; ++u) {
      const int k = k0 + kkh + u;
      int ci, ky, kx;
      if (KHW == 1) { ci = k; ky = 0; kx = 0; }
      else { ci = k / KHW; const int r = k - ci * KHW; ky = r / KW; kx = r - ky * KW; }
      const u16 v = Bin[(unsigned)(ci * inPlane + gbase + ky * inW + kx)];
      if (u & 1) bs.h[u >> 1].y = v; else bs.h[u >> 1].x = v;
    }
  };
  auto commit = [&](int buf) {
    u32x4* ad = (u32x4*)(As[buf] + ar * LDSS + ac);
    ad[0] = a0;
    ad[1] = a1;
    u32x4* bd = (u32x4*)(Bs[buf] + n_local * LDSS + kkh);
    bd[0] = bs.q[0];
    bd[1] = bs.q[1];
  };

  const int NT = K >> 5;   // K is a multiple of 32
  stage(0);
  for (int it = 0; it < NT; ++it) {
    const int buf = it & 1;
    commit(buf);
    if (it + 1 < NT) stage((it + 1) << 5);   // global loads in flight during compute
    __syncthreads();

    const u16* Ab = As[buf];
    const u16* Bb = Bs[buf];
    // A fragment: 16x32 bf16 (lanes 0-15: K 0-15; lanes 16-31: K 16-31)
    FragBF af;
    {
      const int row = wave * 16 + (lane & 15);
      const int kb  = (lane >> 4) * 16;
      af.q[0] = *(const u32x4*)(Ab + row * LDSS + kb);
      af.q[1] = *(const u32x4*)(Ab + row * LDSS + kb + 8);
    }
    // 4 B fragments + WMMA (B 32x16: lanes 0-15 hold K 0-15, lanes 16-31 K 16-31)
    #pragma unroll
    for (int j = 0; j < 4; ++j) {
      FragBF bf;
      const int row = j * 16 + (lane & 15);
      const int kh  = (lane >> 4) * 16;
      bf.q[0] = *(const u32x4*)(Bb + row * LDSS + kh);
      bf.q[1] = *(const u32x4*)(Bb + row * LDSS + kh + 8);
      acc[j] = __builtin_amdgcn_wmma_f32_16x16x32_bf16(
          false, af.v, false, bf.v, (short)0, acc[j], false, false);
    }
    // single barrier per step: next commit targets the other LDS plane
  }

  // ---- epilogue: bias (+ReLU); fp32 flat and/or bf16 (possibly halo-padded) ----
  const int mb = m0 + wave * 16;
  if (outF) {
    #pragma unroll
    for (int v = 0; v < 8; ++v) {
      const int m = mb + v + ((lane >> 4) << 3);
      const float b = bias[m];
      #pragma unroll
      for (int j = 0; j < 4; ++j) {
        const int nn = n0 + j * 16 + (lane & 15);
        float val = acc[j][v] + b;
        if (relu) val = fmaxf(val, 0.f);
        outF[(unsigned)(m * N + nn)] = val;
      }
    }
  }
  if (outB) {
    #pragma unroll
    for (int v = 0; v < 8; ++v) {
      const int m = mb + v + ((lane >> 4) << 3);
      const float b = bias[m];
      #pragma unroll
      for (int j = 0; j < 4; ++j) {
        const int nn = n0 + j * 16 + (lane & 15);
        float val = acc[j][v] + b;
        if (relu) val = fmaxf(val, 0.f);
        const int py = nn >> owShift;
        const int px = nn & ((1 << owShift) - 1);
        outB[(unsigned)(m * obPlane + py * obW + px + obOff)] = f2bf(val);
      }
    }
  }
}

// =====================================================================
// Deconv (kernel == stride): per kernel-position GEMM with disjoint scatter.
// grid.z = S*S kernel positions; A packed per position as [pos][M][K].
// Same halves-staged double-buffered pipeline. IW = 1 << iwShift.
// =====================================================================
__global__ __launch_bounds__(128)
void deconv_gemm_wmma(const u16* __restrict__ Awt, const u16* __restrict__ Bin,
                      const float* __restrict__ bias, float* __restrict__ outF,
                      int M, int K, int N, int iwShift, int S)
{
  __shared__ __align__(16) u16 As[2][64 * LDSS];
  __shared__ __align__(16) u16 Bs[2][64 * LDSS];

  const int t    = threadIdx.x;
  const int lane = t & 31;
  const int wave = t >> 5;
  const int n0   = blockIdx.x * 64;
  const int m0   = blockIdx.y * 64;
  const int pos  = blockIdx.z;
  const int ky   = pos / S, kx = pos - ky * S;
  const u16* A   = Awt + (size_t)pos * M * K;

  const int n_local = t & 63;
  const int kkh     = (t >> 6) * 16;
  const int n  = n0 + n_local;

  const int ar = t >> 1;
  const int ac = (t & 1) * 16;
  const unsigned arow = (unsigned)(m0 + ar) * (unsigned)K + (unsigned)ac;

  v8f acc[4];
  for (int j = 0; j < 4; ++j)
    for (int e = 0; e < 8; ++e) acc[j][e] = 0.f;

  u32x4  a0, a1;
  BStage bs;

  auto stage = [&](int k0) {
    const u32x4* src = (const u32x4*)(A + arow + (unsigned)k0);
    a0 = src[0];
    a1 = src[1];
    #pragma unroll
    for (int u = 0; u < 16; ++u) {
      const u16 v = Bin[(unsigned)((k0 + kkh + u) * N + n)];
      if (u & 1) bs.h[u >> 1].y = v; else bs.h[u >> 1].x = v;
    }
  };
  auto commit = [&](int buf) {
    u32x4* ad = (u32x4*)(As[buf] + ar * LDSS + ac);
    ad[0] = a0;
    ad[1] = a1;
    u32x4* bd = (u32x4*)(Bs[buf] + n_local * LDSS + kkh);
    bd[0] = bs.q[0];
    bd[1] = bs.q[1];
  };

  const int NT = K >> 5;
  stage(0);
  for (int it = 0; it < NT; ++it) {
    const int buf = it & 1;
    commit(buf);
    if (it + 1 < NT) stage((it + 1) << 5);
    __syncthreads();

    const u16* Ab = As[buf];
    const u16* Bb = Bs[buf];
    FragBF af;
    {
      const int row = wave * 16 + (lane & 15);
      const int kb  = (lane >> 4) * 16;
      af.q[0] = *(const u32x4*)(Ab + row * LDSS + kb);
      af.q[1] = *(const u32x4*)(Ab + row * LDSS + kb + 8);
    }
    #pragma unroll
    for (int j = 0; j < 4; ++j) {
      FragBF bf;
      const int row = j * 16 + (lane & 15);
      const int kh  = (lane >> 4) * 16;
      bf.q[0] = *(const u32x4*)(Bb + row * LDSS + kh);
      bf.q[1] = *(const u32x4*)(Bb + row * LDSS + kh + 8);
      acc[j] = __builtin_amdgcn_wmma_f32_16x16x32_bf16(
          false, af.v, false, bf.v, (short)0, acc[j], false, false);
    }
  }

  const int IW    = 1 << iwShift;
  const int OW    = IW * S;
  const int HWout = N * S * S;
  const int mb    = m0 + wave * 16;
  #pragma unroll
  for (int v = 0; v < 8; ++v) {
    const int m = mb + v + ((lane >> 4) << 3);
    const float b = bias[m];
    #pragma unroll
    for (int j = 0; j < 4; ++j) {
      const int nn = n0 + j * 16 + (lane & 15);
      const int iy = nn >> iwShift, ix = nn & (IW - 1);
      outF[(unsigned)(m * HWout + (iy * S + ky) * OW + ix * S + kx)] = acc[j][v] + b;
    }
  }
}

// =====================================================================
// Per-pixel attention fusion over 3 agents, keep ego (row 0 of softmax only):
//   out[c] = sum_m softmax_m(t0.tm / sqrt(C)) * tm[c]
// Input may be halo-padded (plane/inW/padOff); output is flat [C][HW] bf16.
// =====================================================================
__global__ __launch_bounds__(256)
void fuse_attn(const u16* __restrict__ in, u16* __restrict__ out,
               int C, int HW, int plane, int inW, int owShift, int padOff)
{
  const int p = blockIdx.x * blockDim.x + threadIdx.x;
  if (p >= HW) return;
  const int py = p >> owShift;
  const int px = p & ((1 << owShift) - 1);
  const unsigned ib = (unsigned)((py + padOff) * inW + px + padOff);
  const unsigned CH = (unsigned)C * (unsigned)plane;
  float d0 = 0.f, d1 = 0.f, d2 = 0.f;
  for (int c = 0; c < C; ++c) {
    const unsigned o = (unsigned)c * plane + ib;
    const float a = bf2f(in[o]);
    const float b = bf2f(in[CH + o]);
    const float g = bf2f(in[2 * CH + o]);
    d0 += a * a; d1 += a * b; d2 += a * g;
  }
  const float s = rsqrtf((float)C);
  d0 *= s; d1 *= s; d2 *= s;
  const float mx = fmaxf(d0, fmaxf(d1, d2));
  float e0 = __expf(d0 - mx), e1 = __expf(d1 - mx), e2 = __expf(d2 - mx);
  const float inv = 1.f / (e0 + e1 + e2);
  e0 *= inv; e1 *= inv; e2 *= inv;
  for (int c = 0; c < C; ++c) {
    const unsigned o = (unsigned)c * plane + ib;
    const float a = bf2f(in[o]);
    const float b = bf2f(in[CH + o]);
    const float g = bf2f(in[2 * CH + o]);
    out[(unsigned)c * HW + p] = f2bf(e0 * a + e1 * b + e2 * g);
  }
}

// =====================================================================
// EnhanceWeight scaling + fp32 -> bf16 conversion of x (3 imgs) and hist
// (2 imgs), written into HALO-PADDED planes (258x514) with zero borders.
// =====================================================================
#define SRC_W    512
#define SRC_H    256
#define SRC_PL   (SRC_W * SRC_H)         // 131072
#define SRC_IMG  (64 * SRC_PL)           // 8388608
#define PAD_W    (SRC_W + 2)             // 514
#define PAD_PL   (PAD_W * (SRC_H + 2))   // 132612
#define PAD_IMG  (64 * PAD_PL)           // 8487168

__global__ void enhance_cvt(const float* __restrict__ x, const float* __restrict__ hist,
                            const float* __restrict__ td, const float* __restrict__ ewp,
                            const float* __restrict__ ebp,
                            u16* __restrict__ xb, u16* __restrict__ hb)
{
  const float ew = ewp[0], eb = ebp[0];
  const int total  = 5 * PAD_IMG;
  const int stride = gridDim.x * blockDim.x;
  for (int i = blockIdx.x * blockDim.x + threadIdx.x; i < total; i += stride) {
    const int img = i / PAD_IMG;
    const int rem = i - img * PAD_IMG;
    const int c   = rem / PAD_PL;
    const int rr  = rem - c * PAD_PL;
    const int py  = rr / PAD_W;
    const int px  = rr - py * PAD_W;
    u16 o = 0;
    if (py >= 1 && py <= SRC_H && px >= 1 && px <= SRC_W) {
      const int wsel = (img == 0) ? 0 : (img == 1) ? 3 : (img == 2) ? 4 : (img == 3) ? 1 : 2;
      const float w = tanhf(ew * (1.f / (td[wsel] + 0.1f)) + eb) + 1.f;
      const float* src = (img < 3) ? x : hist;
      const int si = ((img < 3) ? img : img - 3) * SRC_IMG + c * SRC_PL + (py - 1) * SRC_W + (px - 1);
      o = f2bf(src[si] * w);
    }
    if (img < 3) xb[img * PAD_IMG + rem] = o;
    else         hb[(img - 3) * PAD_IMG + rem] = o;
  }
}

// ---------- buffer zero-fill (for halo borders of xl0/xl1) ----------
__global__ void zero_u16(u16* __restrict__ p, int n) {
  const int stride = gridDim.x * blockDim.x;
  for (int i = blockIdx.x * blockDim.x + threadIdx.x; i < n; i += stride) p[i] = 0;
}

// ---------- weight prep ----------
__global__ void cvt_bf16(const float* __restrict__ src, u16* __restrict__ dst, int n) {
  const int i = blockIdx.x * blockDim.x + threadIdx.x;
  if (i < n) dst[i] = f2bf(src[i]);
}
// (O,I,S,S) -> [pos][O][I] bf16 (per kernel-position A matrices for deconv GEMMs)
__global__ void pack_deconv(const float* __restrict__ src, u16* __restrict__ dst,
                            int CO, int CI, int S, int total) {
  const int i = blockIdx.x * blockDim.x + threadIdx.x;
  if (i >= total) return;
  const int ci  = i % CI;
  const int t2  = i / CI;
  const int co  = t2 % CO;
  const int pos = t2 / CO;
  const int ky = pos / S, kx = pos - ky * S;
  dst[i] = f2bf(src[(((size_t)co * CI + ci) * S + ky) * S + kx]);
}

// ---------- STA (CBAM channel attention) ----------
__global__ __launch_bounds__(256)
void sta_reduce(const float* __restrict__ x, float* __restrict__ avg,
                float* __restrict__ mxo, int HW)
{
  __shared__ float ssum[256];
  __shared__ float smax[256];
  const int c = blockIdx.x;
  const float* row = x + (size_t)c * HW;
  float s = 0.f, m = -3.402823466e38f;
  for (int i = threadIdx.x; i < HW; i += 256) { const float v = row[i]; s += v; m = fmaxf(m, v); }
  ssum[threadIdx.x] = s; smax[threadIdx.x] = m;
  __syncthreads();
  for (int st = 128; st > 0; st >>= 1) {
    if (threadIdx.x < st) {
      ssum[threadIdx.x] += ssum[threadIdx.x + st];
      smax[threadIdx.x] = fmaxf(smax[threadIdx.x], smax[threadIdx.x + st]);
    }
    __syncthreads();
  }
  if (threadIdx.x == 0) { avg[c] = ssum[0] / (float)HW; mxo[c] = smax[0]; }
}

__global__ void sta_mlp(const float* __restrict__ avg, const float* __restrict__ mxv,
                        const float* __restrict__ w1, const float* __restrict__ w2,
                        float* __restrict__ scale)
{
  __shared__ float ha[32];
  __shared__ float hm[32];
  const int t = threadIdx.x;          // blockDim == 64
  const int h = t & 31;
  const float* z = (t < 32) ? avg : mxv;
  float acc = 0.f;
  for (int c = 0; c < 512; ++c) acc += z[c] * w1[h * 512 + c];
  acc = fmaxf(acc, 0.f);
  if (t < 32) ha[h] = acc; else hm[h] = acc;
  __syncthreads();
  for (int c = t; c < 512; c += 64) {
    float s = 0.f;
    for (int hh = 0; hh < 32; ++hh) s += (ha[hh] + hm[hh]) * w2[c * 32 + hh];
    scale[c] = 1.f / (1.f + __expf(-s));
  }
}

__global__ void sta_scale(float* __restrict__ out, const float* __restrict__ scale, int total)
{
  int i = blockIdx.x * blockDim.x + threadIdx.x;
  const int stride = gridDim.x * blockDim.x;
  for (; i < total; i += stride) out[i] *= scale[i >> 15];  // HW = 32768
}

// =====================================================================
extern "C" void kernel_launch(void* const* d_in, const int* in_sizes, int n_in,
                              void* d_out, int out_size, void* d_ws, size_t ws_size,
                              hipStream_t stream) {
  (void)in_sizes; (void)n_in; (void)out_size; (void)ws_size;

  const float* x    = (const float*)d_in[0];
  const float* hist = (const float*)d_in[1];
  const float* td   = (const float*)d_in[4];
  const float* ew   = (const float*)d_in[6];
  const float* eb   = (const float*)d_in[7];
  const float* b0w  = (const float*)d_in[8];  const float* b0b = (const float*)d_in[9];
  const float* b1w  = (const float*)d_in[10]; const float* b1b = (const float*)d_in[11];
  const float* b2w  = (const float*)d_in[12]; const float* b2b = (const float*)d_in[13];
  const float* d0w  = (const float*)d_in[14]; const float* d0b = (const float*)d_in[15];
  const float* d1w  = (const float*)d_in[16]; const float* d1b = (const float*)d_in[17];
  const float* d2w  = (const float*)d_in[18]; const float* d2b = (const float*)d_in[19];
  const float* w1   = (const float*)d_in[20];
  const float* w2   = (const float*)d_in[21];
  float* out = (float*)d_out;

  const int HW0 = 128 * 256;          // 32768
  const int HW1 = 64 * 128;           // 8192
  const int HW2 = 32 * 64;            // 2048
  // halo-padded planes for conv inputs
  const int PL0 = 130 * 258;          // 33540  (xl0: 128x256 + halo)
  const int PL1 = 66 * 130;           // 8580   (xl1: 64x128 + halo)
  const int XL0N = 3 * 64  * PL0;     // 6,439,680
  const int XL1N = 3 * 128 * PL1;     // 3,294,720

  // ---- workspace carve-out (256B aligned) ----
  char* ws = (char*)d_ws;
  size_t off = 0;
  auto carve = [&](size_t bytes) -> void* {
    size_t a = (off + 255) & ~(size_t)255;
    off = a + bytes;
    return (void*)(ws + a);
  };
  u16* wb0 = (u16*)carve((size_t)64 * 576 * 2);
  u16* wb1 = (u16*)carve((size_t)128 * 576 * 2);
  u16* wb2 = (u16*)carve((size_t)256 * 1152 * 2);
  u16* wd0 = (u16*)carve((size_t)128 * 64 * 2);
  u16* wd1 = (u16*)carve((size_t)4 * 128 * 128 * 2);
  u16* wd2 = (u16*)carve((size_t)16 * 128 * 256 * 2);
  u16* xb  = (u16*)carve((size_t)3 * PAD_IMG * 2);
  u16* hb  = (u16*)carve((size_t)2 * PAD_IMG * 2);
  u16* xl0 = (u16*)carve((size_t)XL0N * 2);
  u16* xl1 = (u16*)carve((size_t)XL1N * 2);
  u16* xl2 = (u16*)carve((size_t)3 * 256 * HW2 * 2);
  u16* f0  = (u16*)carve((size_t)64 * HW0 * 2);
  u16* f1  = (u16*)carve((size_t)128 * HW1 * 2);
  u16* f2  = (u16*)carve((size_t)256 * HW2 * 2);
  float* avg   = (float*)carve(512 * 4);
  float* mxv   = (float*)carve(512 * 4);
  float* scale = (float*)carve(512 * 4);

  const dim3 blk(128);

  // ---- weight conversion / packing + halo zeroing ----
  cvt_bf16<<<(36864 + 255) / 256, 256, 0, stream>>>(b0w, wb0, 36864);
  cvt_bf16<<<(73728 + 255) / 256, 256, 0, stream>>>(b1w, wb1, 73728);
  cvt_bf16<<<(294912 + 255) / 256, 256, 0, stream>>>(b2w, wb2, 294912);
  cvt_bf16<<<(8192 + 255) / 256, 256, 0, stream>>>(d0w, wd0, 8192);
  pack_deconv<<<(65536 + 255) / 256, 256, 0, stream>>>(d1w, wd1, 128, 128, 2, 65536);
  pack_deconv<<<(524288 + 255) / 256, 256, 0, stream>>>(d2w, wd2, 128, 256, 4, 524288);
  zero_u16<<<2048, 256, 0, stream>>>(xl0, XL0N);
  zero_u16<<<2048, 256, 0, stream>>>(xl1, XL1N);

  // ---- enhance + convert into padded planes (borders zeroed in-kernel) ----
  enhance_cvt<<<8192, 256, 0, stream>>>(x, hist, td, ew, eb, xb, hb);

  // ---- level 0 conv (b0) on x -> padded bf16; on hist -> d_out ch 384..511 (fp32, relu) ----
  for (int i = 0; i < 3; ++i)
    conv_gemm_wmma<9, 3, 2><<<dim3(512, 1), blk, 0, stream>>>(
        wb0, xb + (size_t)i * PAD_IMG, b0b, nullptr, xl0 + (size_t)i * 64 * PL0,
        64, 576, HW0, PAD_PL, PAD_W, 8, PL0, 258, 259, 1);
  for (int i = 0; i < 2; ++i)
    conv_gemm_wmma<9, 3, 2><<<dim3(512, 1), blk, 0, stream>>>(
        wb0, hb + (size_t)i * PAD_IMG, b0b, out + (size_t)(384 + i * 64) * HW0, nullptr,
        64, 576, HW0, PAD_PL, PAD_W, 8, 0, 0, 0, 1);

  // ---- fuse level 0 + d0 (1x1, 64->128) -> d_out channels 0..127 ----
  fuse_attn<<<HW0 / 256, 256, 0, stream>>>(xl0, f0, 64, HW0, PL0, 258, 8, 1);
  conv_gemm_wmma<1, 1, 1><<<dim3(512, 2), blk, 0, stream>>>(
      wd0, f0, d0b, out, nullptr, 128, 64, HW0, HW0, 256, 8, 0, 0, 0, 0);

  // ---- level 1 conv (b1) ----
  for (int i = 0; i < 3; ++i)
    conv_gemm_wmma<9, 3, 2><<<dim3(128, 2), blk, 0, stream>>>(
        wb1, xl0 + (size_t)i * 64 * PL0, b1b, nullptr, xl1 + (size_t)i * 128 * PL1,
        128, 576, HW1, PL0, 258, 7, PL1, 130, 131, 1);
  fuse_attn<<<HW1 / 256, 256, 0, stream>>>(xl1, f1, 128, HW1, PL1, 130, 7, 1);
  // d1 deconv 2x2 s2 -> d_out channels 128..255
  deconv_gemm_wmma<<<dim3(128, 2, 4), blk, 0, stream>>>(
      wd1, f1, d1b, out + (size_t)128 * HW0, 128, 128, HW1, 7, 2);

  // ---- level 2 conv (b2) -> flat bf16 (no further conv consumes it) ----
  for (int i = 0; i < 3; ++i)
    conv_gemm_wmma<9, 3, 2><<<dim3(32, 4), blk, 0, stream>>>(
        wb2, xl1 + (size_t)i * 128 * PL1, b2b, nullptr, xl2 + (size_t)i * 256 * HW2,
        256, 1152, HW2, PL1, 130, 6, HW2, 64, 0, 1);
  fuse_attn<<<HW2 / 256, 256, 0, stream>>>(xl2, f2, 256, HW2, HW2, 64, 6, 0);
  // d2 deconv 4x4 s4 -> d_out channels 256..383
  deconv_gemm_wmma<<<dim3(32, 2, 16), blk, 0, stream>>>(
      wd2, f2, d2b, out + (size_t)256 * HW0, 128, 256, HW2, 6, 4);

  // ---- STA channel attention over (512, 32768) ----
  sta_reduce<<<512, 256, 0, stream>>>(out, avg, mxv, HW0);
  sta_mlp<<<1, 64, 0, stream>>>(avg, mxv, w1, w2, scale);
  sta_scale<<<8192, 256, 0, stream>>>(out, scale, 512 * HW0);
}
